// GNNSage_46437186404819
// MI455X (gfx1250) — matrix operand order repaired and checked by reading, not compiled, not run on hardware
//
#include <hip/hip_runtime.h>
#include <hip/hip_bf16.h>

#define NNODES 50000
#define NEDGES 1600000
#define DFEAT  128
#define HID    128
#define NCLS   5
#define BN_EPS 1e-5f

typedef __attribute__((ext_vector_type(16))) _Float16 v16h;
typedef __attribute__((ext_vector_type(8)))  float    v8f;

// ---------------------------------------------------------------------------
// helpers
// ---------------------------------------------------------------------------
__device__ __forceinline__ v16h pack_a(float4 a, float4 b, float4 c, float4 d, float s) {
  // v16h element j (j=0..7)  -> K = k0 + j        (floats a,b)
  // v16h element j (j=8..15) -> K = k0 + j (floats c,d)
  v16h r;
  r[0]  = (_Float16)(a.x*s); r[1]  = (_Float16)(a.y*s); r[2]  = (_Float16)(a.z*s); r[3]  = (_Float16)(a.w*s);
  r[4]  = (_Float16)(b.x*s); r[5]  = (_Float16)(b.y*s); r[6]  = (_Float16)(b.z*s); r[7]  = (_Float16)(b.w*s);
  r[8]  = (_Float16)(c.x*s); r[9]  = (_Float16)(c.y*s); r[10] = (_Float16)(c.z*s); r[11] = (_Float16)(c.w*s);
  r[12] = (_Float16)(d.x*s); r[13] = (_Float16)(d.y*s); r[14] = (_Float16)(d.z*s); r[15] = (_Float16)(d.w*s);
  return r;
}

// ---------------------------------------------------------------------------
// utility kernels
// ---------------------------------------------------------------------------
__global__ void fill_f32(float* __restrict__ p, float v, long n) {
  long i = (long)blockIdx.x * blockDim.x + threadIdx.x;
  long stride = (long)gridDim.x * blockDim.x;
  for (; i < n; i += stride) p[i] = v;
}

// transpose + fp32->f16 weight prep: wt[n*K + k] = w[k*nsrc + n] (zero-pad n >= nsrc)
__global__ void wprep(const float* __restrict__ w, _Float16* __restrict__ wt,
                      int K, int nsrc, int npad) {
  int idx = blockIdx.x * blockDim.x + threadIdx.x;
  if (idx >= npad * K) return;
  int n = idx / K, k = idx - n * K;
  wt[idx] = (n < nsrc) ? (_Float16)w[(long)k * nsrc + n] : (_Float16)0.0f;
}

__global__ void deg_kernel(const int* __restrict__ dst, float* __restrict__ deg, int e) {
  int i = blockIdx.x * blockDim.x + threadIdx.x;
  if (i < e) atomicAdd(&deg[dst[i]], 1.0f);
}

__global__ void rdeg_kernel(const float* __restrict__ deg, float* __restrict__ rdeg, int n) {
  int i = blockIdx.x * blockDim.x + threadIdx.x;
  if (i < n) rdeg[i] = 1.0f / fmaxf(deg[i], 1.0f);
}

// edge-parallel segment-sum: one thread = (edge, 4 consecutive feats)
__global__ void scatter_add_kernel(const float* __restrict__ x, const int* __restrict__ src,
                                   const int* __restrict__ dst, float* __restrict__ acc, long total) {
  long t = (long)blockIdx.x * blockDim.x + threadIdx.x;
  if (t >= total) return;
  int e = (int)(t >> 5);
  int c = ((int)t & 31) << 2;
  const float4 v = *reinterpret_cast<const float4*>(x + (long)src[e] * 128 + c);
  float* a = acc + (long)dst[e] * 128 + c;
  atomicAdd(a + 0, v.x); atomicAdd(a + 1, v.y);
  atomicAdd(a + 2, v.z); atomicAdd(a + 3, v.w);
}

// edge-parallel segment-max of NON-NEGATIVE floats (post-ReLU): uint bit order
// == float order for x >= 0; buffer pre-filled with 0.0f == DGL zero-deg value.
__global__ void scatter_max_kernel(const float* __restrict__ x, const int* __restrict__ src,
                                   const int* __restrict__ dst, float* __restrict__ acc, long total) {
  long t = (long)blockIdx.x * blockDim.x + threadIdx.x;
  if (t >= total) return;
  int e = (int)(t >> 5);
  int c = ((int)t & 31) << 2;
  const float4 v = *reinterpret_cast<const float4*>(x + (long)src[e] * 128 + c);
  unsigned* a = reinterpret_cast<unsigned*>(acc + (long)dst[e] * 128 + c);
  atomicMax(a + 0, __float_as_uint(v.x)); atomicMax(a + 1, __float_as_uint(v.y));
  atomicMax(a + 2, __float_as_uint(v.z)); atomicMax(a + 3, __float_as_uint(v.w));
}

// column sum / sum-of-squares for BatchNorm: sums[0..127]=sum, sums[128..255]=sumsq
__global__ void bn_stats(const float* __restrict__ x, float* __restrict__ sums) {
  int c  = threadIdx.x & 127;
  int r0 = (blockIdx.x << 1) + (threadIdx.x >> 7);
  float s = 0.0f, q = 0.0f;
  for (int r = r0; r < NNODES; r += gridDim.x * 2) {
    float v = x[(long)r * 128 + c];
    s += v; q += v * v;
  }
  atomicAdd(&sums[c], s);
  atomicAdd(&sums[128 + c], q);
}

// coef[0..127]=scale, coef[128..255]=shift
__global__ void bn_final(const float* __restrict__ sums, const float* __restrict__ gamma,
                         const float* __restrict__ beta, float* __restrict__ coef) {
  int c = threadIdx.x;
  float mean = sums[c] * (1.0f / NNODES);
  float var  = sums[128 + c] * (1.0f / NNODES) - mean * mean;   // biased, as torch BN
  float sc   = gamma[c] * rsqrtf(var + BN_EPS);
  coef[c]       = sc;
  coef[128 + c] = beta[c] - mean * sc;
}

// z = relu(bn(hm) + bn(hp)), written in-place into hm
__global__ void bn_add_relu(float* __restrict__ hm, const float* __restrict__ hp,
                            const float* __restrict__ cm, const float* __restrict__ cp, long n) {
  long i = (long)blockIdx.x * blockDim.x + threadIdx.x;
  if (i >= n) return;
  int c = (int)(i & 127);
  float v = cm[c] * hm[i] + cm[128 + c] + cp[c] * hp[i] + cp[128 + c];
  hm[i] = fmaxf(v, 0.0f);
}

// ---------------------------------------------------------------------------
// WMMA GEMM: OUT[row,n] = act( A1 @ W1 (+ (rowScale2 * A2) @ W2) + bias )
// A: [nRowTiles*16, 128] fp32 row-major. Wt: pre-transposed f16, Wt[n*128+k],
// staged cooperatively in LDS (<= 64 KB of the 320 KB/WGP budget).
// One wave owns a 16-row stripe; BOTH full-K A operands held in regs
// (2 x 4 x v16h = 64 VGPRs). Column tiles processed in groups of 2 so only
// 2 accumulators (16 VGPRs) are live at once -> no A re-loads / spills;
// the inner loop reads B exclusively from LDS. 256 thr = 8 waves per block.
// ---------------------------------------------------------------------------
template<bool DUAL, bool RELU, int NTILES, int OUTN>
__global__ __launch_bounds__(256) void sage_gemm(
    const float* __restrict__ A1, const float* __restrict__ A2,
    const float* __restrict__ rowScale2,
    const _Float16* __restrict__ W1t, const _Float16* __restrict__ W2t,
    const float* __restrict__ bias,
    float* __restrict__ OUT, int nRowTiles)
{
  constexpr int WELEMS = NTILES * 16 * 128;          // halves per weight matrix
  constexpr int G = (NTILES >= 2) ? 2 : 1;           // column tiles per group
  __shared__ _Float16 sW1[WELEMS];
  __shared__ _Float16 sW2[DUAL ? WELEMS : 8];

  // cooperative LDS stage of weights (uniform across all waves, before any exit)
  {
    constexpr int NV4 = WELEMS / 8;                  // uint4 (16B) chunks
    const uint4* g1 = reinterpret_cast<const uint4*>(W1t);
    uint4* l1 = reinterpret_cast<uint4*>(sW1);
    #pragma unroll
    for (int i = threadIdx.x; i < NV4; i += 256) l1[i] = g1[i];
    if (DUAL) {
      const uint4* g2 = reinterpret_cast<const uint4*>(W2t);
      uint4* l2 = reinterpret_cast<uint4*>(sW2);
      #pragma unroll
      for (int i = threadIdx.x; i < NV4; i += 256) l2[i] = g2[i];
    }
  }
  __syncthreads();

  int wave = threadIdx.x >> 5;
  int lane = threadIdx.x & 31;
  int tile = blockIdx.x * 8 + wave;
  if (tile >= nRowTiles) return;            // whole wave exits: EXEC all-1 for WMMA
  int r0   = tile << 4;
  int half = lane >> 4;                     // 0: K 0-15 ; 1: K 16-31
  int lr   = lane & 15;
  int row  = r0 + lr;

  const float* pA1 = A1 + (long)row * 128;
  const float* pA2 = nullptr;
  float s2 = 1.0f;
  if (DUAL) {
    pA2 = A2 + (long)row * 128;
    if (rowScale2) s2 = rowScale2[row];     // fused mean-agg division
  }

  v16h a1[4]; v16h a2[4];
  #pragma unroll
  for (int kk = 0; kk < 4; ++kk) {
    int k0 = kk * 32 + half * 16;
    a1[kk] = pack_a(*reinterpret_cast<const float4*>(pA1 + k0),
                    *reinterpret_cast<const float4*>(pA1 + k0 + 4),
                    *reinterpret_cast<const float4*>(pA1 + k0 + 8),
                    *reinterpret_cast<const float4*>(pA1 + k0 + 12), 1.0f);
    if (DUAL)
      a2[kk] = pack_a(*reinterpret_cast<const float4*>(pA2 + k0),
                      *reinterpret_cast<const float4*>(pA2 + k0 + 4),
                      *reinterpret_cast<const float4*>(pA2 + k0 + 8),
                      *reinterpret_cast<const float4*>(pA2 + k0 + 12), s2);
  }

  #pragma unroll
  for (int ng = 0; ng < NTILES; ng += G) {
    v8f acc[G];
    #pragma unroll
    for (int g = 0; g < G; ++g) acc[g] = (v8f){};

    #pragma unroll
    for (int kk = 0; kk < 4; ++kk) {
      #pragma unroll
      for (int g = 0; g < G; ++g) {
        int n = (ng + g) * 16 + lr;         // B column for this lane
        v16h b1 = *reinterpret_cast<const v16h*>(sW1 + n * 128 + half * 16 + kk * 32);
        acc[g] = __builtin_amdgcn_wmma_f32_16x16x32_f16(false, a1[kk], false, b1,
                                                        (short)0, acc[g], false, false);
        if (DUAL) {
          v16h b2 = *reinterpret_cast<const v16h*>(sW2 + n * 128 + half * 16 + kk * 32);
          acc[g] = __builtin_amdgcn_wmma_f32_16x16x32_f16(false, a2[kk], false, b2,
                                                          (short)0, acc[g], false, false);
        }
      }
    }

    #pragma unroll
    for (int g = 0; g < G; ++g) {
      int n = (ng + g) * 16 + lr;
      float bv = (n < OUTN) ? bias[n] : 0.0f;    // folds away when OUTN==128
      #pragma unroll
      for (int r = 0; r < 8; ++r) {              // D VGPR r: row r0+r+half*8, col lr
        int ocol = (ng + g) * 16 + lr;
        if (ocol < OUTN) {
          float v = acc[g][r] + bv;
          if (RELU) v = fmaxf(v, 0.0f);
          OUT[(long)(r0 + half * 8 + r) * OUTN + ocol] = v;
        }
      }
    }
  }
}

// ---------------------------------------------------------------------------
// host launcher
// ---------------------------------------------------------------------------
extern "C" void kernel_launch(void* const* d_in, const int* in_sizes, int n_in,
                              void* d_out, int out_size, void* d_ws, size_t ws_size,
                              hipStream_t stream) {
  const float* features = (const float*)d_in[0];
  const int*   src      = (const int*)  d_in[1];
  const int*   dst      = (const int*)  d_in[2];
  const float* W_self_m = (const float*)d_in[3];
  const float* W_neigh_m= (const float*)d_in[4];
  const float* b_m      = (const float*)d_in[5];
  const float* W_pool   = (const float*)d_in[6];
  const float* b_pool   = (const float*)d_in[7];
  const float* W_self_p = (const float*)d_in[8];
  const float* W_neigh_p= (const float*)d_in[9];
  const float* b_p      = (const float*)d_in[10];
  const float* gamma_m  = (const float*)d_in[11];
  const float* beta_m   = (const float*)d_in[12];
  const float* gamma_p  = (const float*)d_in[13];
  const float* beta_p   = (const float*)d_in[14];
  const float* W_self_o = (const float*)d_in[15];
  const float* W_neigh_o= (const float*)d_in[16];
  const float* b_o      = (const float*)d_in[17];
  float* out = (float*)d_out;

  // ---- carve workspace (all regions re-initialized every call) ----
  char* ws = (char*)d_ws;
  auto carve = [&](size_t bytes) -> char* {
    char* p = ws; ws += (bytes + 255) & ~(size_t)255; return p;
  };
  _Float16* Wt_sm  = (_Float16*)carve(128 * 128 * sizeof(_Float16));
  _Float16* Wt_nm  = (_Float16*)carve(128 * 128 * sizeof(_Float16));
  _Float16* Wt_pl  = (_Float16*)carve(128 * 128 * sizeof(_Float16));
  _Float16* Wt_sp  = (_Float16*)carve(128 * 128 * sizeof(_Float16));
  _Float16* Wt_np  = (_Float16*)carve(128 * 128 * sizeof(_Float16));
  _Float16* Wt_so  = (_Float16*)carve(16  * 128 * sizeof(_Float16));
  _Float16* Wt_no  = (_Float16*)carve(16  * 128 * sizeof(_Float16));
  float* deg   = (float*)carve(NNODES * sizeof(float));
  float* rdeg  = (float*)carve(NNODES * sizeof(float));
  float* sums  = (float*)carve(512 * sizeof(float));   // [0..255] mean-branch, [256..511] pool-branch
  float* bnc   = (float*)carve(512 * sizeof(float));   // scale/shift per branch
  const size_t big = (size_t)NNODES * 128 * sizeof(float);
  float* bufA = (float*)carve(big);   // agg(features) -> agg(z)
  float* bufB = (float*)carve(big);   // h_mean -> z (in place)
  float* bufC = (float*)carve(big);   // P (pool fc) -> h_pool
  float* bufD = (float*)carve(big);   // pooled (segment max)

  const int nRowTiles  = NNODES / 16;                  // 3125 (exact)
  const int gemmBlocks = (nRowTiles + 7) / 8;          // 391
  const long nelB      = (long)NNODES * 128;
  const long totE      = (long)NEDGES * 32;            // edge x 4-float chunks
  const int  scatBlks  = (int)((totE + 255) / 256);

  // ---- weight prep (f16, transposed [n][k]) ----
  wprep<<<64, 256, 0, stream>>>(W_self_m,  Wt_sm, 128, 128, 128);
  wprep<<<64, 256, 0, stream>>>(W_neigh_m, Wt_nm, 128, 128, 128);
  wprep<<<64, 256, 0, stream>>>(W_pool,    Wt_pl, 128, 128, 128);
  wprep<<<64, 256, 0, stream>>>(W_self_p,  Wt_sp, 128, 128, 128);
  wprep<<<64, 256, 0, stream>>>(W_neigh_p, Wt_np, 128, 128, 128);
  wprep<<<8,  256, 0, stream>>>(W_self_o,  Wt_so, 128, NCLS, 16);
  wprep<<<8,  256, 0, stream>>>(W_neigh_o, Wt_no, 128, NCLS, 16);

  // ---- degrees & reciprocal ----
  fill_f32<<<256, 256, 0, stream>>>(deg, 0.0f, NNODES);
  deg_kernel<<<(NEDGES + 255) / 256, 256, 0, stream>>>(dst, deg, NEDGES);
  rdeg_kernel<<<(NNODES + 255) / 256, 256, 0, stream>>>(deg, rdeg, NNODES);

  // ---- mean aggregation of raw features -> bufA ----
  fill_f32<<<2048, 256, 0, stream>>>(bufA, 0.0f, nelB);
  scatter_add_kernel<<<scatBlks, 256, 0, stream>>>(features, src, dst, bufA, totE);

  // ---- layer1 mean branch: h_mean = relu(h@Wsm + (bufA*rdeg)@Wnm + b_m) ----
  sage_gemm<true, true, 8, 128><<<gemmBlocks, 256, 0, stream>>>(
      features, bufA, rdeg, Wt_sm, Wt_nm, b_m, bufB, nRowTiles);

  // ---- pool fc: P = relu(h@W_pool + b_pool) ----
  sage_gemm<false, true, 8, 128><<<gemmBlocks, 256, 0, stream>>>(
      features, nullptr, nullptr, Wt_pl, nullptr, b_pool, bufC, nRowTiles);

  // ---- segment max of P -> bufD ----
  fill_f32<<<2048, 256, 0, stream>>>(bufD, 0.0f, nelB);
  scatter_max_kernel<<<scatBlks, 256, 0, stream>>>(bufC, src, dst, bufD, totE);

  // ---- layer1 pool branch: h_pool = relu(h@Wsp + pooled@Wnp + b_p) -> bufC ----
  sage_gemm<true, true, 8, 128><<<gemmBlocks, 256, 0, stream>>>(
      features, bufD, nullptr, Wt_sp, Wt_np, b_p, bufC, nRowTiles);

  // ---- BatchNorm both branches, add, relu (in place into bufB) ----
  fill_f32<<<2, 256, 0, stream>>>(sums, 0.0f, 512);
  bn_stats<<<1024, 256, 0, stream>>>(bufB, sums);
  bn_stats<<<1024, 256, 0, stream>>>(bufC, sums + 256);
  bn_final<<<1, 128, 0, stream>>>(sums,       gamma_m, beta_m, bnc);
  bn_final<<<1, 128, 0, stream>>>(sums + 256, gamma_p, beta_p, bnc + 256);
  bn_add_relu<<<(int)((nelB + 255) / 256), 256, 0, stream>>>(bufB, bufC, bnc, bnc + 256, nelB);

  // ---- mean aggregation of z -> bufA ----
  fill_f32<<<2048, 256, 0, stream>>>(bufA, 0.0f, nelB);
  scatter_add_kernel<<<scatBlks, 256, 0, stream>>>(bufB, src, dst, bufA, totE);

  // ---- output: logits = z@Wso + (bufA*rdeg)@Wno + b_o  (5 cols, no relu) ----
  sage_gemm<true, false, 1, NCLS><<<gemmBlocks, 256, 0, stream>>>(
      bufB, bufA, rdeg, Wt_so, Wt_no, b_o, out, nRowTiles);
}
